// Decoder_80315888436037
// MI455X (gfx1250) — matrix operand order repaired and checked by reading, not compiled, not run on hardware
//
#include <hip/hip_runtime.h>
#include <hip/hip_bf16.h>
#include <math.h>

// ---------------------------------------------------------------- types
typedef __bf16 bf16;
typedef bf16  v8bf  __attribute__((ext_vector_type(8)));
typedef bf16  v16bf __attribute__((ext_vector_type(16)));
typedef float v8f   __attribute__((ext_vector_type(8)));

static constexpr int Bn = 32;     // batch
static constexpr int Sn = 128;    // source positions
static constexpr int Tn = 64;     // target steps
static constexpr int Vn = 32000;  // vocab
static constexpr int En = 256;    // embed dim
static constexpr int Hn = 512;    // hidden
static constexpr int H2 = 2 * Hn;          // 1024
static constexpr int Kbig = En + Hn + H2;  // 1792  (z = [x | s | ctx])
static constexpr int Nbig = H2 + Hn;       // 1536  (gates | s_tilde pre-act)

__device__ __forceinline__ bf16 to_bf16(float f) { return (bf16)f; }

// ---------------------------------------------------------------- generic bf16 WMMA GEMM
// C[M,N] = A[M,K](bf16,row-major) * Bt[N,K](bf16, i.e. B transposed) + bias_scale*bias[N]
// Each wave computes a 16x64 strip: one A-fragment reused across 4 N-tiles
// (4 back-to-back WMMAs sharing the A operand). Block = 4 waves = 16(M) x 256(N).
// All dims divide exactly (N % 256 == 0, M % 16 == 0, K % 32 == 0): EXEC stays all-ones.
__global__ __launch_bounds__(128) void k_gemm_wmma(
    const bf16* __restrict__ A, int lda,
    const bf16* __restrict__ Bt, int ldb,
    const float* __restrict__ bias, float bias_scale,
    float* __restrict__ C, int ldc, int M, int N, int K) {
  const int wave = threadIdx.x >> 5;
  const int lane = threadIdx.x & 31;
  const int n0 = (blockIdx.x * 16 + wave * 4) * 16;  // this wave: cols n0..n0+63
  const int m0 = blockIdx.y * 16;
  if (n0 >= N || m0 >= M) return;   // wave-uniform exit
  const int half = lane >> 4;
  const int lidx = lane & 15;

  const bf16* ap = A  + (size_t)(m0 + lidx) * lda + half * 16;  // A rows striped by M
  const bf16* bp = Bt + (size_t)(n0 + lidx) * ldb + half * 16;  // B cols striped by N
  const size_t bstr = (size_t)16 * ldb;                         // next N-tile

  v8f acc[4] = {{}, {}, {}, {}};
  for (int k0 = 0; k0 < K; k0 += 32) {
    __builtin_prefetch((const void*)(ap + k0 + 32), 0, 1);
    __builtin_prefetch((const void*)(bp + k0 + 32), 0, 1);
    __builtin_prefetch((const void*)(bp + 3 * bstr + k0 + 32), 0, 1);
    v8bf alo = *(const v8bf*)(ap + k0);        // K = k0 + half*16 + [0..7]
    v8bf ahi = *(const v8bf*)(ap + k0 + 8);    // K = k0 + half*16 + 8 + [0..7]
    v16bf afr;
#pragma unroll
    for (int i = 0; i < 8; ++i) { afr[i] = alo[i]; afr[i + 8] = ahi[i]; }
    v16bf b0 = *(const v16bf*)(bp + k0);
    v16bf b1 = *(const v16bf*)(bp + bstr + k0);
    v16bf b2 = *(const v16bf*)(bp + 2 * bstr + k0);
    v16bf b3 = *(const v16bf*)(bp + 3 * bstr + k0);
    acc[0] = __builtin_amdgcn_wmma_f32_16x16x32_bf16(false, afr, false, b0, (short)0, acc[0], false, false);
    acc[1] = __builtin_amdgcn_wmma_f32_16x16x32_bf16(false, afr, false, b1, (short)0, acc[1], false, false);
    acc[2] = __builtin_amdgcn_wmma_f32_16x16x32_bf16(false, afr, false, b2, (short)0, acc[2], false, false);
    acc[3] = __builtin_amdgcn_wmma_f32_16x16x32_bf16(false, afr, false, b3, (short)0, acc[3], false, false);
  }

#pragma unroll
  for (int j = 0; j < 4; ++j) {
    const int col = n0 + j * 16 + lidx;
    const float bv = bias ? bias_scale * bias[col] : 0.0f;
#pragma unroll
    for (int v = 0; v < 8; ++v) {
      const int row = m0 + half * 8 + v;   // C: lane holds col, rows m0+8*half+v
      C[(size_t)row * ldc + col] = acc[j][v] + bv;
    }
  }
}

// ---------------------------------------------------------------- prep kernels
__global__ void k_cvt(const float* __restrict__ x, bf16* __restrict__ y, int n) {
  int i = blockIdx.x * blockDim.x + threadIdx.x;
  if (i < n) y[i] = to_bf16(x[i]);
}

// Wt[n*K + k] = W[k*N + n]   (W is [K,N] row-major -> Bt layout [N][K])
__global__ void k_cvt_transpose(const float* __restrict__ W, bf16* __restrict__ Wt,
                                int K, int N) {
  int i = blockIdx.x * blockDim.x + threadIdx.x;
  if (i >= N * K) return;
  int n = i / K, k = i - n * K;
  Wt[i] = to_bf16(W[(size_t)k * N + n]);
}

__global__ void k_gather_embed(const int* __restrict__ target,
                               const float* __restrict__ embed,
                               bf16* __restrict__ tgt_emb) {
  int i = blockIdx.x * blockDim.x + threadIdx.x;   // over B*T*E
  if (i >= Bn * Tn * En) return;
  int bt = i / En, e = i - bt * En;
  tgt_emb[i] = to_bf16(embed[(size_t)target[bt] * En + e]);
}

// W_big^T [Nbig][Kbig]: rows 0..1023 = gate outputs, rows 1024..1535 = s_tilde outputs
// K: 0..255 from x (Wi_g/Wi), 256..767 from prev_s (Wh_g/Wh), 768..1791 from ctx (Wc_g/Wc)
__global__ void k_build_wbig(const float* __restrict__ Wi_g, const float* __restrict__ Wh_g,
                             const float* __restrict__ Wc_g, const float* __restrict__ Wi,
                             const float* __restrict__ Wh, const float* __restrict__ Wc,
                             bf16* __restrict__ Wt) {
  int i = blockIdx.x * blockDim.x + threadIdx.x;
  if (i >= Nbig * Kbig) return;
  int n = i / Kbig, k = i - n * Kbig;
  float v;
  if (n < H2) {
    if (k < En)            v = Wi_g[(size_t)k * H2 + n];
    else if (k < En + Hn)  v = Wh_g[(size_t)(k - En) * H2 + n];
    else                   v = Wc_g[(size_t)(k - En - Hn) * H2 + n];
  } else {
    int n2 = n - H2;
    if (k < En)            v = Wi[(size_t)k * Hn + n2];
    else if (k < En + Hn)  v = Wh[(size_t)(k - En) * Hn + n2];
    else                   v = Wc[(size_t)(k - En - Hn) * Hn + n2];
  }
  Wt[i] = to_bf16(v);
}

__global__ void k_build_bbig(const float* bi_g, const float* bh_g, const float* bc_g,
                             const float* bi, const float* bh, const float* bc,
                             float* __restrict__ bbig) {
  int n = blockIdx.x * blockDim.x + threadIdx.x;
  if (n >= Nbig) return;
  bbig[n] = (n < H2) ? (bi_g[n] + bh_g[n] + bc_g[n])
                     : (bi[n - H2] + bh[n - H2] + bc[n - H2]);
}

__global__ void k_copy_f32(const float* __restrict__ x, float* __restrict__ y, int n) {
  int i = blockIdx.x * blockDim.x + threadIdx.x;
  if (i < n) y[i] = x[i];
}

// ---------------------------------------------------------------- per-step kernels
// score[b,s] = sum_h tanh(enc_proj[b,s,h] + q[b,h]) * W_att[h] + b_att ; one wave per (b,s)
__global__ __launch_bounds__(128) void k_attn_score(
    const float* __restrict__ enc_proj, const float* __restrict__ q,
    const float* __restrict__ W_att, const float* __restrict__ b_att,
    float* __restrict__ score) {
  int pair = blockIdx.x * 4 + (threadIdx.x >> 5);   // b*S + s
  int lane = threadIdx.x & 31;
  const float* ep = enc_proj + (size_t)pair * Hn;
  const float* qp = q + (size_t)(pair >> 7) * Hn;   // b = pair / S
  float acc = 0.0f;
  for (int j = lane; j < Hn; j += 32)
    acc += tanhf(ep[j] + qp[j]) * W_att[j];
#pragma unroll
  for (int o = 16; o > 0; o >>= 1) acc += __shfl_down(acc, o, 32);
  if (lane == 0) score[pair] = acc + b_att[0];
}

__global__ __launch_bounds__(128) void k_softmax(float* __restrict__ score) {
  __shared__ float red[Sn];
  int b = blockIdx.x, s = threadIdx.x;
  float v = score[b * Sn + s];
  red[s] = v; __syncthreads();
  for (int o = Sn / 2; o > 0; o >>= 1) {
    if (s < o) red[s] = fmaxf(red[s], red[s + o]);
    __syncthreads();
  }
  float m = red[0]; __syncthreads();
  float e = expf(v - m);
  red[s] = e; __syncthreads();
  for (int o = Sn / 2; o > 0; o >>= 1) {
    if (s < o) red[s] += red[s + o];
    __syncthreads();
  }
  score[b * Sn + s] = e / red[0];
}

// ctx[b,j] = sum_s alpha[b,s] * hiddens[b,s,j]   (j over 2H)
__global__ void k_ctx(const float* __restrict__ alpha, const float* __restrict__ hiddens,
                      float* __restrict__ ctx) {
  int i = blockIdx.x * blockDim.x + threadIdx.x;   // over B*2H
  if (i >= Bn * H2) return;
  int b = i / H2, j = i - b * H2;
  const float* hp = hiddens + (size_t)b * Sn * H2 + j;
  const float* ap = alpha + b * Sn;
  float acc = 0.0f;
  for (int s = 0; s < Sn; ++s) acc += ap[s] * hp[(size_t)s * H2];
  ctx[i] = acc;
}

// z[b, 0:256]=x_t(bf16), z[b,256:768]=prev_s, z[b,768:1792]=ctx
__global__ void k_pack_z(const bf16* __restrict__ tgt_emb, const float* __restrict__ prev_s,
                         const float* __restrict__ ctx, bf16* __restrict__ z, int t) {
  int i = blockIdx.x * blockDim.x + threadIdx.x;   // over B*Kbig
  if (i >= Bn * Kbig) return;
  int b = i / Kbig, k = i - b * Kbig;
  bf16 v;
  if (k < En)            v = tgt_emb[((size_t)b * Tn + t) * En + k];
  else if (k < En + Hn)  v = to_bf16(prev_s[b * Hn + (k - En)]);
  else                   v = to_bf16(ctx[b * H2 + (k - En - Hn)]);
  z[i] = v;
}

// r = sigmoid(g[:, :H]); s~ = tanh(g[:, 1024:1536]); s' = (1-r)*s + r*s~
__global__ void k_cell_update(const float* __restrict__ gates, float* __restrict__ prev_s,
                              float* __restrict__ dec_h, int t) {
  int i = blockIdx.x * blockDim.x + threadIdx.x;   // over B*H
  if (i >= Bn * Hn) return;
  int b = i / Hn, h = i - b * Hn;
  float r  = 1.0f / (1.0f + expf(-gates[(size_t)b * Nbig + h]));
  float st = tanhf(gates[(size_t)b * Nbig + H2 + h]);
  float ns = (1.0f - r) * prev_s[i] + r * st;
  prev_s[i] = ns;
  dec_h[((size_t)b * Tn + t) * Hn + h] = ns;
}

// A2[b*T+t, h] = bf16(dec_h + 0.5*sentiment[b,h])   (folds style logits into final GEMM)
__global__ void k_pack_final(const float* __restrict__ dec_h, const float* __restrict__ sent,
                             bf16* __restrict__ A2) {
  int i = blockIdx.x * blockDim.x + threadIdx.x;   // over B*T*H
  if (i >= Bn * Tn * Hn) return;
  int row = i / Hn, h = i - row * Hn;
  int b = row / Tn;
  A2[i] = to_bf16(dec_h[i] + 0.5f * sent[b * Hn + h]);
}

// ---------------------------------------------------------------- host orchestration
extern "C" void kernel_launch(void* const* d_in, const int* in_sizes, int n_in,
                              void* d_out, int out_size, void* d_ws, size_t ws_size,
                              hipStream_t stream) {
  (void)in_sizes; (void)n_in; (void)out_size; (void)ws_size;
  const float* content   = (const float*)d_in[0];
  const float* sentiment = (const float*)d_in[1];
  const float* hiddens   = (const float*)d_in[2];
  const int*   target    = (const int*)d_in[3];
  const float* embed     = (const float*)d_in[4];
  const float* W_enc     = (const float*)d_in[5];
  const float* b_enc     = (const float*)d_in[6];
  const float* W_prev    = (const float*)d_in[7];
  const float* b_prev    = (const float*)d_in[8];
  const float* W_att     = (const float*)d_in[9];
  const float* b_att     = (const float*)d_in[10];
  const float* Wi_g = (const float*)d_in[11];
  const float* bi_g = (const float*)d_in[12];
  const float* Wh_g = (const float*)d_in[13];
  const float* bh_g = (const float*)d_in[14];
  const float* Wc_g = (const float*)d_in[15];
  const float* bc_g = (const float*)d_in[16];
  const float* Wi   = (const float*)d_in[17];
  const float* bi   = (const float*)d_in[18];
  const float* Wh   = (const float*)d_in[19];
  const float* bh   = (const float*)d_in[20];
  const float* Wc   = (const float*)d_in[21];
  const float* bc   = (const float*)d_in[22];
  const float* W_out= (const float*)d_in[23];
  const float* b_out= (const float*)d_in[24];
  float* out = (float*)d_out;

  // ---- workspace carve-up (deterministic offsets)
  char* ws = (char*)d_ws;
  size_t off = 0;
  auto carve = [&](size_t bytes) -> char* {
    char* p = ws + off;
    off += (bytes + 255) & ~(size_t)255;
    return p;
  };
  bf16*  tgt_emb  = (bf16*) carve((size_t)Bn * Tn * En * 2);       // 1 MB
  bf16*  hid_bf   = (bf16*) carve((size_t)Bn * Sn * H2 * 2);       // 8 MB
  bf16*  Wenc_t   = (bf16*) carve((size_t)Hn * H2 * 2);            // 1 MB
  bf16*  Wprev_t  = (bf16*) carve((size_t)Hn * Hn * 2);            // 0.5 MB
  bf16*  Wbig_t   = (bf16*) carve((size_t)Nbig * Kbig * 2);        // 5.5 MB
  float* bbig     = (float*)carve((size_t)Nbig * 4);
  bf16*  Wout_t   = (bf16*) carve((size_t)Vn * Hn * 2);            // 32.8 MB
  float* enc_proj = (float*)carve((size_t)Bn * Sn * Hn * 4);       // 8 MB
  float* q        = (float*)carve((size_t)Bn * Hn * 4);
  float* alpha    = (float*)carve((size_t)Bn * Sn * 4);
  float* ctx      = (float*)carve((size_t)Bn * H2 * 4);
  bf16*  zb       = (bf16*) carve((size_t)Bn * Kbig * 2);
  float* gates    = (float*)carve((size_t)Bn * Nbig * 4);
  float* prev_s   = (float*)carve((size_t)Bn * Hn * 4);
  bf16*  s_bf     = (bf16*) carve((size_t)Bn * Hn * 2);
  float* dec_h    = (float*)carve((size_t)Bn * Tn * Hn * 4);       // 4 MB
  bf16*  A2       = (bf16*) carve((size_t)Bn * Tn * Hn * 2);       // 2 MB

  // ---- one-time prep (re-done every call: deterministic, no cross-call state)
  k_gather_embed<<<(Bn * Tn * En + 255) / 256, 256, 0, stream>>>(target, embed, tgt_emb);
  k_cvt<<<(Bn * Sn * H2 + 255) / 256, 256, 0, stream>>>(hiddens, hid_bf, Bn * Sn * H2);
  k_cvt_transpose<<<(Hn * H2 + 255) / 256, 256, 0, stream>>>(W_enc, Wenc_t, H2, Hn);
  k_cvt_transpose<<<(Hn * Hn + 255) / 256, 256, 0, stream>>>(W_prev, Wprev_t, Hn, Hn);
  k_cvt_transpose<<<((size_t)Vn * Hn + 255) / 256, 256, 0, stream>>>(W_out, Wout_t, Hn, Vn);
  k_build_wbig<<<(Nbig * Kbig + 255) / 256, 256, 0, stream>>>(Wi_g, Wh_g, Wc_g, Wi, Wh, Wc, Wbig_t);
  k_build_bbig<<<(Nbig + 255) / 256, 256, 0, stream>>>(bi_g, bh_g, bc_g, bi, bh, bc, bbig);
  k_copy_f32<<<(Bn * Hn + 255) / 256, 256, 0, stream>>>(content, prev_s, Bn * Hn);

  // enc_proj = hiddens @ W_enc + b_enc    [4096,1024]x[1024,512]
  k_gemm_wmma<<<dim3(Hn / 256, (Bn * Sn) / 16), 128, 0, stream>>>(
      hid_bf, H2, Wenc_t, H2, b_enc, 1.0f, enc_proj, Hn, Bn * Sn, Hn, H2);

  // ---- sequential scan over T (stream order carries the dependency)
  for (int t = 0; t < Tn; ++t) {
    k_cvt<<<(Bn * Hn + 255) / 256, 256, 0, stream>>>(prev_s, s_bf, Bn * Hn);
    // q = prev_s @ W_prev + b_prev   [32,512]x[512,512]
    k_gemm_wmma<<<dim3(Hn / 256, Bn / 16), 128, 0, stream>>>(
        s_bf, Hn, Wprev_t, Hn, b_prev, 1.0f, q, Hn, Bn, Hn, Hn);
    k_attn_score<<<(Bn * Sn) / 4, 128, 0, stream>>>(enc_proj, q, W_att, b_att, alpha);
    k_softmax<<<Bn, Sn, 0, stream>>>(alpha);
    k_ctx<<<(Bn * H2 + 255) / 256, 256, 0, stream>>>(alpha, hiddens, ctx);
    k_pack_z<<<(Bn * Kbig + 255) / 256, 256, 0, stream>>>(tgt_emb, prev_s, ctx, zb, t);
    // gates|s~ = z @ W_big + b_big   [32,1792]x[1792,1536]
    k_gemm_wmma<<<dim3(Nbig / 256, Bn / 16), 128, 0, stream>>>(
        zb, Kbig, Wbig_t, Kbig, bbig, 1.0f, gates, Nbig, Bn, Nbig, Kbig);
    k_cell_update<<<(Bn * Hn + 255) / 256, 256, 0, stream>>>(gates, prev_s, dec_h, t);
  }

  // ---- output projection with style fold: (dec_h + 0.5*sent) @ W_out + 1.5*b_out
  k_pack_final<<<(Bn * Tn * Hn + 255) / 256, 256, 0, stream>>>(dec_h, sentiment, A2);
  k_gemm_wmma<<<dim3(Vn / 256, (Bn * Tn) / 16), 128, 0, stream>>>(
      A2, Hn, Wout_t, Hn, b_out, 1.5f, out, Vn, Bn * Tn, Vn, Hn);
}
